// AudioEncoder_90254442758309
// MI455X (gfx1250) — matrix operand order repaired and checked by reading, not compiled, not run on hardware
//
#include <hip/hip_runtime.h>
#include <hip/hip_bf16.h>
#include <math.h>

typedef __bf16 bf16_t;
typedef __attribute__((ext_vector_type(8)))  __bf16 v8bf;
typedef __attribute__((ext_vector_type(16))) __bf16 v16bf;
typedef __attribute__((ext_vector_type(8)))  float  v8f;

// Problem constants
constexpr int BATCH  = 16;
constexpr int SLEN   = 524288;
constexpr int NFFT   = 2048;
constexpr int HOP    = 512;
constexpr int NMEL   = 128;
constexpr int T1     = 1025;
constexpr int T2     = 513;
constexpr int VOCAB  = 4096;
constexpr int DMODEL = 512;
constexpr int NROWS  = BATCH * T2;   // 8208

__device__ inline v16bf cat16(v8bf lo, v8bf hi) {
    return __builtin_shufflevector(lo, hi, 0, 1, 2, 3, 4, 5, 6, 7,
                                           8, 9, 10, 11, 12, 13, 14, 15);
}

// Load a 16-element A/B fragment from two aligned 16B slices.
__device__ inline v16bf ldfrag(const bf16_t* p) {
    v8bf lo = *(const v8bf*)(p);
    v8bf hi = *(const v8bf*)(p + 8);
    return cat16(lo, hi);
}

// ---------------- preprocessing kernels ----------------

__global__ void cast_f32_to_bf16(const float* __restrict__ s, bf16_t* __restrict__ d, int n) {
    int i = blockIdx.x * 256 + threadIdx.x;
    if (i < n) d[i] = (bf16_t)s[i];
}

// w (Cout, Cin, 3) f32 -> Aperm[oc][dk*Cin + ci] bf16 (K-reordered GEMM A)
__global__ void pack_weights(const float* __restrict__ w, bf16_t* __restrict__ a,
                             int Cout, int Cin, int log2Cin) {
    int idx = blockIdx.x * 256 + threadIdx.x;
    int Ktot = 3 * Cin;
    if (idx < Cout * Ktot) {
        int oc = idx / Ktot;
        int kp = idx - oc * Ktot;
        int dk = kp >> log2Cin;
        int ci = kp & (Cin - 1);
        a[idx] = (bf16_t)w[(size_t)(oc * Cin + ci) * 3 + dk];
    }
}

__global__ void cb_norms(const float* __restrict__ cb, float* __restrict__ cn) {
    int n = blockIdx.x * 256 + threadIdx.x;
    if (n < VOCAB) {
        float s = 0.f;
        const float* row = cb + (size_t)n * DMODEL;
        for (int k = 0; k < DMODEL; ++k) s = fmaf(row[k], row[k], s);
        cn[n] = s;
    }
}

__global__ void init_scores(unsigned long long* __restrict__ s, int n) {
    int i = blockIdx.x * 256 + threadIdx.x;
    if (i < n) s[i] = ~0ULL;
}

// ---------------- mel-log spectrogram ----------------

__global__ __launch_bounds__(256)
void mel_kernel(const float* __restrict__ wav, const float* __restrict__ fb,
                bf16_t* __restrict__ mel) {
    __shared__ float sre[NFFT];
    __shared__ float sim[NFFT];
    const int t   = blockIdx.x;
    const int b   = blockIdx.y;
    const int tid = threadIdx.x;
    const float* wb = wav + (size_t)b * SLEN;

    for (int j = tid; j < NFFT; j += 256) {
        int p = t * HOP + j - (NFFT / 2);
        if (p < 0) p = -p;
        if (p >= SLEN) p = 2 * SLEN - 2 - p;
        float win = 0.5f - 0.5f * __cosf(6.28318530718f * (float)j / (float)NFFT);
        int rev = (int)(__brev((unsigned)j) >> 21);
        sre[rev] = wb[p] * win;
        sim[rev] = 0.f;
    }
    for (int s = 1; s <= 11; ++s) {
        __syncthreads();
        const int half = 1 << (s - 1);
        const int m    = half << 1;
        for (int i = tid; i < NFFT / 2; i += 256) {
            int g  = i >> (s - 1);
            int p  = i & (half - 1);
            int i1 = g * m + p;
            int i2 = i1 + half;
            float ang = -6.28318530718f * (float)p / (float)m;
            float sw, cw;
            __sincosf(ang, &sw, &cw);
            float re2 = sre[i2], im2 = sim[i2];
            float tr = cw * re2 - sw * im2;
            float ti = cw * im2 + sw * re2;
            float re1 = sre[i1], im1 = sim[i1];
            sre[i1] = re1 + tr; sim[i1] = im1 + ti;
            sre[i2] = re1 - tr; sim[i2] = im1 - ti;
        }
    }
    __syncthreads();
    for (int k = tid; k <= NFFT / 2; k += 256) {
        float re = sre[k], im = sim[k];
        sim[k] = re * re + im * im;
    }
    __syncthreads();
    if (tid < NMEL) {
        float acc = 0.f;
        for (int k = 0; k <= NFFT / 2; ++k)
            acc = fmaf(sim[k], fb[k * NMEL + tid], acc);
        acc = __logf(fmaxf(acc, 1e-5f));
        mel[((size_t)b * NMEL + tid) * T1 + t] = (bf16_t)acc;
    }
}

// ---------------- conv1d(k=3)+GELU as WMMA GEMM, LDS-staged ----------------
// Block = 256 threads (8 waves): 128 outch x (NT*16) time. Activation slice
// staged once in LDS as [tloc][ci] (+8 pad). K' = dk*Cin+ci ordering makes
// every 32-chunk single-dk -> branch-free frags. A (weights) double-buffered
// from global; B = 2x ds_load_b128 per frag from LDS.

template <int NT>
__global__ __launch_bounds__(256)
void conv_gemm_kernel(const bf16_t* __restrict__ x, const bf16_t* __restrict__ w,
                      const float* __restrict__ bias,
                      bf16_t* __restrict__ y, bf16_t* __restrict__ yT,
                      int Cin, int log2Cin, int Tin, int Cout, int Tout,
                      int stride, int writeT) {
    extern __shared__ bf16_t tile[];
    const int tid  = threadIdx.x;
    const int wv   = tid >> 5;
    const int lane = tid & 31;
    const int hs   = lane >> 4;
    const int r    = lane & 15;
    const int tB   = blockIdx.x * (NT * 16);
    const int b    = blockIdx.z;
    const int Ktot = 3 * Cin;
    const int CP   = Cin + 8;
    const int TLOC = NT * 16 * stride + 2;
    const bf16_t* xb = x + (size_t)b * Cin * Tin;

    // stage activation slice (zero-padded edges)
    for (int ci = wv; ci < Cin; ci += 8) {
        const bf16_t* src = xb + (size_t)ci * Tin;
        for (int tl = lane; tl < TLOC; tl += 32) {
            int tin = tB * stride - 1 + tl;
            float v = (tin >= 0 && tin < Tin) ? (float)src[tin] : 0.f;
            tile[tl * CP + ci] = (bf16_t)v;
        }
    }
    __syncthreads();

    const int ocB = blockIdx.y * 128 + wv * 16;
    const bf16_t* wrow = w + (size_t)(ocB + r) * Ktot;

    v8f acc[NT] = {};
    v16bf a = ldfrag(wrow + hs * 8);                 // low16 at hs*8, high at +16
    {
        v8bf hi = *(const v8bf*)(wrow + 16 + hs * 8);
        a = cat16(*(const v8bf*)(wrow + hs * 8), hi);
    }
    for (int kc = 0; kc < Ktot; kc += 32) {
        v16bf an = a;
        if (kc + 32 < Ktot) {                        // double-buffer next A
            const bf16_t* wn = wrow + kc + 32;
            an = cat16(*(const v8bf*)(wn + hs * 8), *(const v8bf*)(wn + 16 + hs * 8));
        }
        const int dk  = kc >> log2Cin;
        const int cib = kc & (Cin - 1);
#pragma unroll
        for (int j = 0; j < NT; ++j) {
            int tloc = (j * 16 + r) * stride + dk;
            const bf16_t* base = tile + tloc * CP + cib + hs * 8;
            v16bf bm = cat16(*(const v8bf*)(base), *(const v8bf*)(base + 16));
            acc[j] = __builtin_amdgcn_wmma_f32_16x16x32_bf16(
                false, a, false, bm, (short)0, acc[j], false, false);
        }
        a = an;
    }

    // bias + exact GELU + store
#pragma unroll
    for (int j = 0; j < NT; ++j) {
        const int t = tB + j * 16 + r;
        if (writeT) {
            v8bf pack;
#pragma unroll
            for (int i = 0; i < 8; ++i) {
                int oc = ocB + hs * 8 + i;
                float v = acc[j][i] + bias[oc];
                v = 0.5f * v * (1.0f + erff(v * 0.70710678118f));
                pack[i] = (bf16_t)v;
            }
            if (t < Tout)                             // one b128 store per lane
                *(v8bf*)(yT + ((size_t)b * Tout + t) * Cout + ocB + hs * 8) = pack;
        } else {
#pragma unroll
            for (int i = 0; i < 8; ++i) {
                int oc = ocB + hs * 8 + i;
                float v = acc[j][i] + bias[oc];
                v = 0.5f * v * (1.0f + erff(v * 0.70710678118f));
                if (t < Tout)
                    y[((size_t)b * Cout + oc) * Tout + t] = (bf16_t)v;
            }
        }
    }
}

// ---------------- VQ: argmin_n (||c_n||^2 - 2 x.c_n) via WMMA ----------------
// One wave per 32(rows) x 64(codes) tile: 2 A-frags x 4 B-frags -> 8 WMMAs
// per K-chunk, B (codebook, L2-resident) loaded once per chunk. Explicit
// double-buffering across the 16 K-chunks hides load latency.

__device__ inline unsigned fkey(float f) {
    unsigned u = __float_as_uint(f);
    return (u & 0x80000000u) ? ~u : (u | 0x80000000u);
}

__global__ __launch_bounds__(32)
void vq_kernel(const bf16_t* __restrict__ flat, const bf16_t* __restrict__ cbB,
               const float* __restrict__ cn, unsigned long long* __restrict__ scores) {
    const int lane = threadIdx.x & 31;
    const int hs   = lane >> 4;
    const int r    = lane & 15;
    const int nB   = blockIdx.x * 64;
    const int mB   = blockIdx.y * 32;
    const bf16_t* arow0 = flat + (size_t)(mB + r) * DMODEL;
    const bf16_t* arow1 = flat + (size_t)(mB + 16 + r) * DMODEL;

    v8f acc[2][4] = {};
    // current fragments (chunk 0)
    v16bf a0 = cat16(*(const v8bf*)(arow0 + hs * 8), *(const v8bf*)(arow0 + 16 + hs * 8));
    v16bf a1 = cat16(*(const v8bf*)(arow1 + hs * 8), *(const v8bf*)(arow1 + 16 + hs * 8));
    v16bf bc[4];
#pragma unroll
    for (int j = 0; j < 4; ++j) {
        const bf16_t* brow = cbB + (size_t)(nB + j * 16 + r) * DMODEL + hs * 8;
        bc[j] = cat16(*(const v8bf*)(brow), *(const v8bf*)(brow + 16));
    }
#pragma unroll
    for (int kc = 0; kc < DMODEL - 32; kc += 32) {
        const int kn = kc + 32;
        v16bf a0n = cat16(*(const v8bf*)(arow0 + kn + hs * 8),
                          *(const v8bf*)(arow0 + kn + 16 + hs * 8));
        v16bf a1n = cat16(*(const v8bf*)(arow1 + kn + hs * 8),
                          *(const v8bf*)(arow1 + kn + 16 + hs * 8));
        v16bf bn[4];
#pragma unroll
        for (int j = 0; j < 4; ++j) {
            const bf16_t* brow = cbB + (size_t)(nB + j * 16 + r) * DMODEL + kn + hs * 8;
            bn[j] = cat16(*(const v8bf*)(brow), *(const v8bf*)(brow + 16));
        }
#pragma unroll
        for (int j = 0; j < 4; ++j)
            acc[0][j] = __builtin_amdgcn_wmma_f32_16x16x32_bf16(
                false, a0, false, bc[j], (short)0, acc[0][j], false, false);
#pragma unroll
        for (int j = 0; j < 4; ++j)
            acc[1][j] = __builtin_amdgcn_wmma_f32_16x16x32_bf16(
                false, a1, false, bc[j], (short)0, acc[1][j], false, false);
        a0 = a0n; a1 = a1n;
#pragma unroll
        for (int j = 0; j < 4; ++j) bc[j] = bn[j];
    }
#pragma unroll
    for (int j = 0; j < 4; ++j)        // last chunk
        acc[0][j] = __builtin_amdgcn_wmma_f32_16x16x32_bf16(
            false, a0, false, bc[j], (short)0, acc[0][j], false, false);
#pragma unroll
    for (int j = 0; j < 4; ++j)
        acc[1][j] = __builtin_amdgcn_wmma_f32_16x16x32_bf16(
            false, a1, false, bc[j], (short)0, acc[1][j], false, false);

#pragma unroll
    for (int s = 0; s < 2; ++s) {
#pragma unroll
        for (int i = 0; i < 8; ++i) {
            unsigned key = 0xFFFFFFFFu, idx = 0xFFFFFFFFu;
#pragma unroll
            for (int j = 0; j < 4; ++j) {
                unsigned nIdx = (unsigned)(nB + j * 16 + r);
                float score = cn[nIdx] - 2.0f * acc[s][j][i];
                unsigned k = fkey(score);
                if (k < key || (k == key && nIdx < idx)) { key = k; idx = nIdx; }
            }
            for (int m = 1; m < 16; m <<= 1) {   // stays within 16-lane half
                unsigned ok = __shfl_xor(key, m, 32);
                unsigned oi = __shfl_xor(idx, m, 32);
                if (ok < key || (ok == key && oi < idx)) { key = ok; idx = oi; }
            }
            int row = mB + s * 16 + hs * 8 + i;
            if (r == 0 && row < NROWS)
                atomicMin(&scores[row],
                          ((unsigned long long)key << 32) | (unsigned long long)idx);
        }
    }
}

// ---------------- decode tokens + gather embeddings ----------------

__global__ __launch_bounds__(128)
void finalize_kernel(const unsigned long long* __restrict__ scores,
                     const float* __restrict__ cb,
                     float* __restrict__ emb, float* __restrict__ tok) {
    int row = blockIdx.x;
    unsigned token = (unsigned)(scores[row] & 0xFFFFFFFFULL);
    const float* src = cb + (size_t)token * DMODEL;
    float* dst = emb + (size_t)row * DMODEL;
    for (int c = threadIdx.x; c < DMODEL; c += 128) dst[c] = src[c];
    if (threadIdx.x == 0) tok[row] = (float)token;
}

// ---------------- host orchestration ----------------

extern "C" void kernel_launch(void* const* d_in, const int* in_sizes, int n_in,
                              void* d_out, int out_size, void* d_ws, size_t ws_size,
                              hipStream_t stream) {
    const float* wav = (const float*)d_in[0];
    const float* fb  = (const float*)d_in[1];
    const float* w1  = (const float*)d_in[2];
    const float* b1  = (const float*)d_in[3];
    const float* w2  = (const float*)d_in[4];
    const float* b2  = (const float*)d_in[5];
    const float* w3  = (const float*)d_in[6];
    const float* b3  = (const float*)d_in[7];
    const float* cb  = (const float*)d_in[8];

    char* ws = (char*)d_ws;
    size_t off = 0;
    auto carve = [&](size_t bytes) -> char* {
        char* p = ws + off;
        off = (off + bytes + 255) & ~(size_t)255;
        return p;
    };
    bf16_t* A1   = (bf16_t*)carve((size_t)256 * 384 * 2);
    bf16_t* A2   = (bf16_t*)carve((size_t)512 * 768 * 2);
    bf16_t* A3   = (bf16_t*)carve((size_t)512 * 1536 * 2);
    bf16_t* cbB  = (bf16_t*)carve((size_t)VOCAB * DMODEL * 2);
    float*  cn   = (float*) carve((size_t)VOCAB * 4);
    bf16_t* mel  = (bf16_t*)carve((size_t)BATCH * NMEL * T1 * 2);
    bf16_t* h1   = (bf16_t*)carve((size_t)BATCH * 256 * T1 * 2);
    bf16_t* h2   = (bf16_t*)carve((size_t)BATCH * 512 * T2 * 2);
    bf16_t* h3T  = (bf16_t*)carve((size_t)BATCH * T2 * 512 * 2);
    unsigned long long* scores = (unsigned long long*)carve((size_t)NROWS * 8);

    pack_weights<<<(256 * 384 + 255) / 256, 256, 0, stream>>>(w1, A1, 256, 128, 7);
    pack_weights<<<(512 * 768 + 255) / 256, 256, 0, stream>>>(w2, A2, 512, 256, 8);
    pack_weights<<<(512 * 1536 + 255) / 256, 256, 0, stream>>>(w3, A3, 512, 512, 9);
    cast_f32_to_bf16<<<(VOCAB * DMODEL + 255) / 256, 256, 0, stream>>>(cb, cbB, VOCAB * DMODEL);
    cb_norms<<<(VOCAB + 255) / 256, 256, 0, stream>>>(cb, cn);

    mel_kernel<<<dim3(T1, BATCH), 256, 0, stream>>>(wav, fb, mel);

    // conv1: 128->256, s1, NT=4: LDS = 66*136*2 = 17,952 B
    conv_gemm_kernel<4><<<dim3((T1 + 63) / 64, 256 / 128, BATCH), 256,
                          (size_t)(4 * 16 * 1 + 2) * (128 + 8) * 2, stream>>>(
        mel, A1, b1, h1, nullptr, 128, 7, T1, 256, T1, 1, 0);
    // conv2: 256->512, s2, NT=2: LDS = 66*264*2 = 34,848 B
    conv_gemm_kernel<2><<<dim3((T2 + 31) / 32, 512 / 128, BATCH), 256,
                          (size_t)(2 * 16 * 2 + 2) * (256 + 8) * 2, stream>>>(
        h1, A2, b2, h2, nullptr, 256, 8, T1, 512, T2, 2, 0);
    // conv3: 512->512, s1, NT=2, transposed out: LDS = 34*520*2 = 35,360 B
    conv_gemm_kernel<2><<<dim3((T2 + 31) / 32, 512 / 128, BATCH), 256,
                          (size_t)(2 * 16 * 1 + 2) * (512 + 8) * 2, stream>>>(
        h2, A3, b3, nullptr, h3T, 512, 9, T2, 512, T2, 1, 1);

    init_scores<<<(NROWS + 255) / 256, 256, 0, stream>>>(scores, NROWS);
    vq_kernel<<<dim3(VOCAB / 64, (NROWS + 31) / 32), 32, 0, stream>>>(h3T, cbB, cn, scores);

    float* emb = (float*)d_out;
    float* tok = (float*)d_out + (size_t)NROWS * DMODEL;
    finalize_kernel<<<NROWS, 128, 0, stream>>>(scores, cb, emb, tok);
}
